// ScaledDotProductAttention_40750649704986
// MI455X (gfx1250) — matrix operand order repaired and checked by reading, not compiled, not run on hardware
//
#include <hip/hip_runtime.h>

// ---------- types ----------
typedef __bf16       v16bf __attribute__((ext_vector_type(16)));
typedef float        v8f   __attribute__((ext_vector_type(8)));
typedef unsigned int u32x4 __attribute__((ext_vector_type(4)));
typedef int          i32x8 __attribute__((ext_vector_type(8)));

__device__ __forceinline__ __bf16 f32_to_bf16(float f) {
    unsigned u = __builtin_bit_cast(unsigned, f);
    u += 0x7FFFu + ((u >> 16) & 1u);            // round-to-nearest-even
    unsigned short h = (unsigned short)(u >> 16);
    return __builtin_bit_cast(__bf16, h);
}

// ---------- TDM: DMA a [128 rows x 32 bf16] tile (row pitch = pitch_elems)
// from global memory into LDS with +16B padding per 64B row -> LDS pitch 40 bf16.
// Descriptor layout per CDNA5 ISA sec 8.3/8.4 (2D tensor, groups 0-1 only).
__device__ __forceinline__ void tdm_load_tile_32x128(
        const __bf16* gptr, unsigned lds_off, unsigned pitch_elems) {
    unsigned long long ga = (unsigned long long)(uintptr_t)gptr;
    u32x4 g0;
    g0[0] = 1u;                                       // count=1, user mode
    g0[1] = lds_off;                                  // lds_addr (bytes)
    g0[2] = (unsigned)ga;                             // global_addr[31:0]
    g0[3] = (unsigned)((ga >> 32) & 0x1FFFFFFu)       // global_addr[56:32]
          | (2u << 30);                               // type = 2 ("image")
    i32x8 g1;
    g1[0] = (1 << 16)                                 // data_size = 2 bytes
          | (1 << 20)                                 // pad_enable
          | (3 << 22)                                 // pad_interval: 16 DWORDs (64B row)
          | (3 << 25);                                // pad_amount: 4 DWORDs (16B)
    g1[1] = (int)(32u << 16);                         // tensor_dim0 = 32 (lo16 in hi half)
    g1[2] = (int)(128u << 16);                        // tensor_dim0 hi=0 | tensor_dim1 = 128
    g1[3] = (int)(32u << 16);                         // tensor_dim1 hi=0 | tile_dim0 = 32
    g1[4] = 128;                                      // tile_dim1 = 128, tile_dim2 = 0
    g1[5] = (int)pitch_elems;                         // tensor_dim0_stride[31:0]
    g1[6] = 0;                                        // stride0 hi | tensor_dim1_stride lo
    g1[7] = 0;
    asm volatile("tensor_load_to_lds %0, %1" :: "s"(g0), "s"(g1) : "memory");
}

// ---------- fp32 -> bf16 convert ----------
__global__ __launch_bounds__(256) void convert_f32_bf16(
        const float* __restrict__ src, __bf16* __restrict__ dst, long long n) {
    long long i = (long long)blockIdx.x * blockDim.x + threadIdx.x;
    long long stride = (long long)gridDim.x * blockDim.x;
    for (; i < n; i += stride) dst[i] = f32_to_bf16(src[i]);
}

// ---------- generic bf16 WMMA GEMM (TDM-staged, double buffered) ----------
// Block tile 128(M) x 128(N), 8 waves (256 thr), wave tile 32x64, K-step 32.
// BMODE 0: B source row-major [K][N]   (weights W[din][dout], V[key][dout])
// BMODE 1: B source row-major [N][K]   (K-matrix: scores B = K^T) -> TDM for B too
// EPI 0: bf16 out, +bias[n]    EPI 1: f32 out, *scale (causal block skip)
// EPI 2: f32 out
template <int BMODE, int EPI>
__global__ __launch_bounds__(256) void gemm_bf16_wmma(
        const __bf16* __restrict__ A, const __bf16* __restrict__ Bsrc,
        void* __restrict__ Cout, const float* __restrict__ bias,
        int lda, int ldb, int ldc, int Kdim,
        long long aBatch, long long bBatch, long long cBatch, float scale) {
    const int blockN0 = blockIdx.x * 128;
    const int blockM0 = blockIdx.y * 128;
    if (EPI == 1 && blockN0 > blockM0 + 127) return;   // fully-masked causal tile

    const int bz = blockIdx.z;
    A    += (long long)bz * aBatch;
    Bsrc += (long long)bz * bBatch;
    float*  Cf = (float*)Cout  + (long long)bz * cBatch;
    __bf16* Cb = (__bf16*)Cout + (long long)bz * cBatch;

    __shared__ __bf16 As[2][128][40];   // [buf][m][k]  (TDM-filled, 40 = 32 + pad)
    __shared__ __bf16 Bs[2][128][40];   // [buf][n][k]  (N-major)

    const int tid   = threadIdx.x;
    const int lane  = tid & 31;
    const int wave  = tid >> 5;
    const int waveM = wave >> 1;     // 0..3  -> rows  waveM*32
    const int waveN = wave & 1;      // 0..1  -> cols  waveN*64

    const unsigned asOff0 = (unsigned)(uintptr_t)&As[0][0][0];
    const unsigned asOff1 = (unsigned)(uintptr_t)&As[1][0][0];
    const unsigned bsOff0 = (unsigned)(uintptr_t)&Bs[0][0][0];
    const unsigned bsOff1 = (unsigned)(uintptr_t)&Bs[1][0][0];

    v8f acc[2][4];
    #pragma unroll
    for (int mi = 0; mi < 2; ++mi)
        #pragma unroll
        for (int ni = 0; ni < 4; ++ni)
            #pragma unroll
            for (int r = 0; r < 8; ++r) acc[mi][ni][r] = 0.0f;

    // manual transposed staging of B for BMODE 0 ([K][N] source)
    auto stageB0 = [&](int k0, int buf) {
        const int k    = tid >> 3;            // 0..31
        const int nseg = (tid & 7) * 16;
        const __bf16* src = Bsrc + (long long)(k0 + k) * ldb + blockN0 + nseg;
        #pragma unroll
        for (int i = 0; i < 16; ++i) Bs[buf][nseg + i][k] = src[i];
    };
    auto issueTdm = [&](int k0, int buf) {    // wave-0 only, scalar path
        tdm_load_tile_32x128(A + (long long)blockM0 * lda + k0,
                             buf ? asOff1 : asOff0, (unsigned)lda);
        if (BMODE == 1)
            tdm_load_tile_32x128(Bsrc + (long long)blockN0 * ldb + k0,
                                 buf ? bsOff1 : bsOff0, (unsigned)ldb);
    };

    // ---- prologue: fill buffer 0 ----
    if (wave == 0) issueTdm(0, 0);
    if (BMODE == 0) stageB0(0, 0);

    const int nIter = Kdim >> 5;
    for (int it = 0; it < nIter; ++it) {
        const int p       = it & 1;
        const bool hasNext = (it + 1) < nIter;

        // issue next tile into the other buffer (overlaps with compute below)
        if (hasNext) {
            if (wave == 0) issueTdm((it + 1) << 5, p ^ 1);
            if (BMODE == 0) stageB0((it + 1) << 5, p ^ 1);
        }
        if (wave == 0) {
            if (hasNext) __builtin_amdgcn_s_wait_tensorcnt(BMODE == 1 ? 2 : 1);
            else         __builtin_amdgcn_s_wait_tensorcnt(0);
        }
        __syncthreads();   // buffer p ready for everyone

        // ---- fragments + WMMA on buffer p ----
        const int mrowBase = waveM * 32 + (lane & 15);
        const int khA = (lane >> 4) * 8;     // A: lanes 16-31 hold K 8-15 / 24-31
        v16bf afrag[2];
        #pragma unroll
        for (int mi = 0; mi < 2; ++mi) {
            const int mrow = mrowBase + mi * 16;
            #pragma unroll
            for (int v = 0; v < 8; ++v) {
                const int kb = ((v < 4) ? (2 * v) : (16 + 2 * (v - 4))) + khA;
                afrag[mi][2 * v]     = As[p][mrow][kb];
                afrag[mi][2 * v + 1] = As[p][mrow][kb + 1];
            }
        }
        const int khB = (lane >> 4) * 8;     // B: lanes 16-31 hold K 8-15 / 24-31
        #pragma unroll
        for (int ni = 0; ni < 4; ++ni) {
            const int ncol = waveN * 64 + ni * 16 + (lane & 15);
            v16bf bfrag;
            #pragma unroll
            for (int v = 0; v < 8; ++v) {
                const int kbB = ((v < 4) ? (2 * v) : (16 + 2 * (v - 4))) + khB;
                bfrag[2 * v]     = Bs[p][ncol][kbB];
                bfrag[2 * v + 1] = Bs[p][ncol][kbB + 1];
            }
            acc[0][ni] = __builtin_amdgcn_wmma_f32_16x16x32_bf16(
                false, afrag[0], false, bfrag, (short)0, acc[0][ni], false, false);
            acc[1][ni] = __builtin_amdgcn_wmma_f32_16x16x32_bf16(
                false, afrag[1], false, bfrag, (short)0, acc[1][ni], false, false);
        }
        __syncthreads();   // reads of buffer p done -> safe to refill next iter
    }

    // ---- epilogue (C/D layout: VGPR r -> M = r + 8*(lane>=16), N = lane&15) ----
    const int mAdd = 8 * (lane >> 4);
    const int nCol = lane & 15;
    #pragma unroll
    for (int mi = 0; mi < 2; ++mi) {
        #pragma unroll
        for (int ni = 0; ni < 4; ++ni) {
            const int nBase = blockN0 + waveN * 64 + ni * 16 + nCol;
            #pragma unroll
            for (int r = 0; r < 8; ++r) {
                const int m = blockM0 + waveM * 32 + mi * 16 + r + mAdd;
                const long long idx = (long long)m * ldc + nBase;
                float v = acc[mi][ni][r];
                if (EPI == 0)      Cb[idx] = f32_to_bf16(v + bias[nBase]);
                else if (EPI == 1) Cf[idx] = v * scale;
                else               Cf[idx] = v;
            }
        }
    }
}

// ---------- causal row max / sumexp ----------
__global__ __launch_bounds__(256) void row_stats(
        const float* __restrict__ S, float* __restrict__ stats) {
    const int row = blockIdx.x;              // 0..8191 (= b*2048 + q)
    const int q   = row & 2047;
    const float* p = S + (long long)row * 2048;
    const int len = q + 1;
    __shared__ float red[256];
    const int tid = threadIdx.x;

    float m = -3.0e38f;
    for (int k = tid; k < len; k += 256) m = fmaxf(m, p[k]);
    red[tid] = m; __syncthreads();
    for (int off = 128; off > 0; off >>= 1) {
        if (tid < off) red[tid] = fmaxf(red[tid], red[tid + off]);
        __syncthreads();
    }
    m = red[0]; __syncthreads();

    float s = 0.0f;
    for (int k = tid; k < len; k += 256) s += __expf(p[k] - m);
    red[tid] = s; __syncthreads();
    for (int off = 128; off > 0; off >>= 1) {
        if (tid < off) red[tid] += red[tid + off];
        __syncthreads();
    }
    if (tid == 0) { stats[2 * row] = m; stats[2 * row + 1] = red[0]; }
}

// ---------- normalize to bf16 probabilities (causal zeros) ----------
__global__ __launch_bounds__(256) void normalize_probs(
        const float* __restrict__ S, const float* __restrict__ stats,
        __bf16* __restrict__ P, long long total) {
    long long i = (long long)blockIdx.x * blockDim.x + threadIdx.x;
    long long stride = (long long)gridDim.x * blockDim.x;
    for (; i < total; i += stride) {
        const int k   = (int)(i & 2047);
        const long long row = i >> 11;           // b*2048 + q
        const int q   = (int)(row & 2047);
        float out = 0.0f;
        if (k <= q) {
            const float m = stats[2 * row];
            const float l = stats[2 * row + 1];
            out = __expf(S[i] - m) * __frcp_rn(l);
        }
        P[i] = f32_to_bf16(out);
    }
}

// ---------- launch ----------
extern "C" void kernel_launch(void* const* d_in, const int* in_sizes, int n_in,
                              void* d_out, int out_size, void* d_ws, size_t ws_size,
                              hipStream_t stream) {
    (void)in_sizes; (void)n_in; (void)out_size; (void)ws_size;
    const float* x   = (const float*)d_in[0];
    const float* w_q = (const float*)d_in[1];
    const float* b_q = (const float*)d_in[2];
    const float* w_k = (const float*)d_in[3];
    const float* b_k = (const float*)d_in[4];
    const float* w_v = (const float*)d_in[5];
    const float* b_v = (const float*)d_in[6];

    const long long B = 4, Sq = 2048, D = 1024;
    const long long TOK = B * Sq;            // 8192
    const long long XN  = TOK * D;           // 8,388,608
    const long long WN  = D * D;             // 1,048,576
    const long long SN  = B * Sq * Sq;       // 16,777,216

    char* ws = (char*)d_ws;
    __bf16* xb    = (__bf16*)ws;                 ws += XN * 2;
    __bf16* wb    = (__bf16*)ws;                 ws += 3 * WN * 2;
    __bf16* Qb    = (__bf16*)ws;                 ws += XN * 2;
    __bf16* Kb    = (__bf16*)ws;                 ws += XN * 2;
    __bf16* Vb    = (__bf16*)ws;                 ws += XN * 2;
    float*  Sraw  = (float*)ws;                  ws += SN * 4;
    __bf16* Pb    = (__bf16*)ws;                 ws += SN * 2;
    float*  stats = (float*)ws;

    // 1) convert inputs to bf16
    convert_f32_bf16<<<1024, 256, 0, stream>>>(x,   xb,          XN);
    convert_f32_bf16<<<256,  256, 0, stream>>>(w_q, wb + 0 * WN, WN);
    convert_f32_bf16<<<256,  256, 0, stream>>>(w_k, wb + 1 * WN, WN);
    convert_f32_bf16<<<256,  256, 0, stream>>>(w_v, wb + 2 * WN, WN);

    // 2) Q/K/V projections: [8192x1024] = Xb * W + b   -> bf16
    dim3 gProj(D / 128, TOK / 128, 1);
    gemm_bf16_wmma<0, 0><<<gProj, 256, 0, stream>>>(xb, wb + 0 * WN, Qb, b_q,
        (int)D, (int)D, (int)D, (int)D, 0, 0, 0, 1.0f);
    gemm_bf16_wmma<0, 0><<<gProj, 256, 0, stream>>>(xb, wb + 1 * WN, Kb, b_k,
        (int)D, (int)D, (int)D, (int)D, 0, 0, 0, 1.0f);
    gemm_bf16_wmma<0, 0><<<gProj, 256, 0, stream>>>(xb, wb + 2 * WN, Vb, b_v,
        (int)D, (int)D, (int)D, (int)D, 0, 0, 0, 1.0f);

    // 3) scores = Q * K^T / sqrt(D)  (per batch, causal tiles skipped, TDM both tiles)
    dim3 gScore(Sq / 128, Sq / 128, B);
    gemm_bf16_wmma<1, 1><<<gScore, 256, 0, stream>>>(Qb, Kb, Sraw, nullptr,
        (int)D, (int)D, (int)Sq, (int)D,
        Sq * D, Sq * D, Sq * Sq, 0.03125f);

    // 4) softmax stats + normalize -> bf16 probs
    row_stats<<<(int)(B * Sq), 256, 0, stream>>>(Sraw, stats);
    normalize_probs<<<4096, 256, 0, stream>>>(Sraw, stats, Pb, SN);

    // 5) O = P * V  (per batch) -> fp32 d_out
    dim3 gPV(D / 128, Sq / 128, B);
    gemm_bf16_wmma<0, 2><<<gPV, 256, 0, stream>>>(Pb, Vb, d_out, nullptr,
        (int)Sq, (int)D, (int)D, (int)Sq,
        Sq * Sq, Sq * D, Sq * D, 1.0f);
}
